// ChebConvNet_6090263626253
// MI455X (gfx1250) — matrix operand order repaired and checked by reading, not compiled, not run on hardware
//
#include <hip/hip_runtime.h>
#include <hip/hip_bf16.h>
#include <math.h>

#define CH          128
#define NUM_GRAPHS  64
#define D_HID       32
#define D_OUT       16
#define KTILES      12          // K = 3*128 = 384 = 12 * 32
#define CTILES      8           // 128 cols = 8 * 16

typedef __attribute__((ext_vector_type(16))) __bf16 v16bf;
typedef __attribute__((ext_vector_type(8)))  float  v8f;

// ---------------------------------------------------------------- utilities
__global__ void zero_f32(float* __restrict__ p, int n) {
  int i = blockIdx.x * blockDim.x + threadIdx.x;
  if (i < n) p[i] = 0.f;
}

// deg[s] += 1 for every non-self-loop edge
__global__ void deg_kernel(const int* __restrict__ src, const int* __restrict__ dst,
                           float* __restrict__ deg, int E) {
  int e = blockIdx.x * blockDim.x + threadIdx.x;
  if (e >= E) return;
  int s = src[e], d = dst[e];
  if (s != d) atomicAdd(&deg[s], 1.0f);
}

// in-place deg -> d^{-1/2}
__global__ void dinv_kernel(float* __restrict__ deg, int n) {
  int i = blockIdx.x * blockDim.x + threadIdx.x;
  if (i >= n) return;
  float dv = deg[i];
  deg[i] = (dv > 0.f) ? rsqrtf(dv) : 0.f;
}

// w[e] = -dinv[src]*dinv[dst] for non-self-loops else 0
__global__ void ew_kernel(const int* __restrict__ src, const int* __restrict__ dst,
                          const float* __restrict__ dinv, float* __restrict__ w, int E) {
  int e = blockIdx.x * blockDim.x + threadIdx.x;
  if (e >= E) return;
  int s = src[e], d = dst[e];
  w[e] = (s != d) ? (-dinv[s] * dinv[d]) : 0.f;
}

// out[dst] += w * v[src]  (32 lanes per edge, float4 per lane)
__global__ void spmv_scatter(const float* __restrict__ v, const int* __restrict__ src,
                             const int* __restrict__ dst, const float* __restrict__ w,
                             float* __restrict__ out, int E) {
  int t = blockIdx.x * blockDim.x + threadIdx.x;
  int e = t >> 5;
  if (e >= E) return;
  float we = w[e];
  if (we == 0.f) return;
  int c = (t & 31) * 4;
  int s = src[e], d = dst[e];
  const float4 xv = *(const float4*)(v + (size_t)s * CH + c);
  float* o = out + (size_t)d * CH + c;
  atomicAdd(o + 0, we * xv.x);
  atomicAdd(o + 1, we * xv.y);
  atomicAdd(o + 2, we * xv.z);
  atomicAdd(o + 3, we * xv.w);
}

// tx2 = 2*tx2 - tx0   (Chebyshev recurrence, in place)
__global__ void cheb_t2_kernel(float* __restrict__ tx2, const float* __restrict__ tx0, int n) {
  int i = blockIdx.x * blockDim.x + threadIdx.x;
  if (i < n) tx2[i] = 2.f * tx2[i] - tx0[i];
}

// ---------------------------------------------------------------- W packing
// Pack (384,128) f32 weight into bf16, B-fragment-major:
//   Wp[((t*8 + c)*32 + lane)*16 + j] = W[(t*32 + kB(lane) + j + (j>=8 ? 8 : 0))*128 + c*16 + (lane&15)]
// so one lane's 16-element WMMA B fragment is one contiguous 32-byte run.
__global__ void pack_w_bf16(const float* __restrict__ W, __bf16* __restrict__ Wp) {
  int i = blockIdx.x * blockDim.x + threadIdx.x;
  if (i >= KTILES * CTILES * 32 * 16) return;
  int j    = i & 15;
  int lane = (i >> 4) & 31;
  int c    = (i >> 9) & 7;
  int t    = i >> 12;
  int n    = c * 16 + (lane & 15);
  int kB   = (lane < 16) ? 0 : 8;           // matches A: k = 8h+j, 16+8h+(j-8)
  int K    = t * 32 + kB + j + (j >= 8 ? 8 : 0);
  Wp[i] = (__bf16)W[(size_t)K * CH + n];
}

// ---------------------------------------------------------------- WMMA GEMM
// out[N,128] = [Tx0|Tx1|Tx2] (N,384) @ Wcat (384,128) + bias, optional SiLU.
// One wave owns a 16-row x 128-col strip: 1 A-fragment load feeds 8 WMMAs.
// K loop fully unrolled; per k-tile all 8 B fragments are loaded before the
// WMMA burst so load latency overlaps matrix ops.
__global__ void __launch_bounds__(256)
cheb_gemm_wmma(const float* __restrict__ Tx0, const float* __restrict__ Tx1,
               const float* __restrict__ Tx2, const __bf16* __restrict__ Wp,
               const float* __restrict__ bias, float* __restrict__ out,
               int ntiles, int act) {
  const int lane    = threadIdx.x & 31;
  const int wave    = threadIdx.x >> 5;
  const int rowtile = blockIdx.x * 8 + wave;     // wave-uniform
  if (rowtile >= ntiles) return;                 // EXEC stays all-ones in live waves
  const int rowbase = rowtile * 16;
  const int half    = (lane < 16) ? 0 : 1;
  const int mrow    = lane & 15;

  const int kA  = half ? 8  : 0;                 // A frag K run 1 (ISA 7.12.2)
  const int kA2 = half ? 24 : 16;                // A frag K run 2

  v8f acc[CTILES];
#pragma unroll
  for (int c = 0; c < CTILES; ++c) acc[c] = v8f{0.f,0.f,0.f,0.f,0.f,0.f,0.f,0.f};

  const float* row0 = Tx0 + (size_t)(rowbase + mrow) * CH;
  const float* row1 = Tx1 + (size_t)(rowbase + mrow) * CH;
  const float* row2 = Tx2 + (size_t)(rowbase + mrow) * CH;

#pragma unroll
  for (int t = 0; t < KTILES; ++t) {
    // compile-time buffer select -> SGPR base + immediate offsets
    const float* pa = (t < 4 ? row0 : (t < 8 ? row1 : row2)) + (t & 3) * 32;

    v16bf afrag;
#pragma unroll
    for (int v = 0; v < 4; ++v) {                // VGPRs 0..3: K = kA + 2v, +1
      float2 f = *(const float2*)(pa + kA + 2 * v);
      afrag[2 * v]     = (__bf16)f.x;
      afrag[2 * v + 1] = (__bf16)f.y;
    }
#pragma unroll
    for (int v = 0; v < 4; ++v) {                // VGPRs 4..7: K = kA2 + 2v, +1
      float2 f = *(const float2*)(pa + kA2 + 2 * v);
      afrag[8 + 2 * v]     = (__bf16)f.x;
      afrag[8 + 2 * v + 1] = (__bf16)f.y;
    }

    // load all 8 B fragments for this k-tile, then burst the 8 WMMAs
    const __bf16* wp = Wp + (((size_t)t * CTILES * 32) + lane) * 16;
    v16bf bf[CTILES];
#pragma unroll
    for (int c = 0; c < CTILES; ++c)
      bf[c] = *(const v16bf*)(wp + (size_t)c * 32 * 16);
#pragma unroll
    for (int c = 0; c < CTILES; ++c)
      acc[c] = __builtin_amdgcn_wmma_f32_16x16x32_bf16(
          false, afrag, false, bf[c], (short)0, acc[c], false, false);
  }

#pragma unroll
  for (int c = 0; c < CTILES; ++c) {
    const int col = c * 16 + (lane & 15);
    const float bv = bias[col];
#pragma unroll
    for (int v = 0; v < 8; ++v) {                // C/D layout: M = v + 8*half
      int m = rowbase + v + half * 8;
      float val = acc[c][v] + bv;
      if (act) val = val / (1.f + __expf(-val)); // SiLU
      out[(size_t)m * CH + col] = val;
    }
  }
}

// ---------------------------------------------------------------- pool + MLP
__global__ void pool_kernel(const float* __restrict__ h, const int* __restrict__ batch,
                            float* __restrict__ g, int n) {
  int t = blockIdx.x * blockDim.x + threadIdx.x;
  int node = t >> 5;
  if (node >= n) return;
  int c = (t & 31) * 4;
  int b = batch[node];
  const float4 v = *(const float4*)(h + (size_t)node * CH + c);
  float* o = g + (size_t)b * CH + c;
  atomicAdd(o + 0, v.x);
  atomicAdd(o + 1, v.y);
  atomicAdd(o + 2, v.z);
  atomicAdd(o + 3, v.w);
}

__global__ void __launch_bounds__(512)
mlp_kernel(const float* __restrict__ g, const float* __restrict__ P1,
           const float* __restrict__ pb1, const float* __restrict__ P2,
           const float* __restrict__ pb2, float* __restrict__ out) {
  __shared__ float sg[NUM_GRAPHS * CH];
  __shared__ float st[NUM_GRAPHS * D_HID];
  int tid = threadIdx.x;
  for (int i = tid; i < NUM_GRAPHS * CH; i += 512) sg[i] = g[i];
  __syncthreads();
  for (int i = tid; i < NUM_GRAPHS * D_HID; i += 512) {
    int m = i / D_HID, nn = i % D_HID;
    float s = pb1[nn];
    for (int k = 0; k < CH; ++k) s += sg[m * CH + k] * P1[k * D_HID + nn];
    st[i] = fmaxf(s, 0.f);
  }
  __syncthreads();
  for (int i = tid; i < NUM_GRAPHS * D_OUT; i += 512) {
    int m = i / D_OUT, nn = i % D_OUT;
    float s = pb2[nn];
    for (int k = 0; k < D_HID; ++k) s += st[m * D_HID + k] * P2[k * D_OUT + nn];
    out[i] = s;
  }
}

// ---------------------------------------------------------------- launch
extern "C" void kernel_launch(void* const* d_in, const int* in_sizes, int n_in,
                              void* d_out, int out_size, void* d_ws, size_t ws_size,
                              hipStream_t stream) {
  const float* x    = (const float*)d_in[0];
  const int*   eidx = (const int*)  d_in[1];   // [2, E]
  const int*   batc = (const int*)  d_in[2];
  const float* W1   = (const float*)d_in[3];   // (3,128,128) == (384,128)
  const float* b1   = (const float*)d_in[4];
  const float* Wh   = (const float*)d_in[5];   // (3,3,128,128)
  const float* bh   = (const float*)d_in[6];   // (3,128)
  const float* W2   = (const float*)d_in[7];
  const float* b2   = (const float*)d_in[8];
  const float* P1   = (const float*)d_in[9];
  const float* pb1  = (const float*)d_in[10];
  const float* P2   = (const float*)d_in[11];
  const float* pb2  = (const float*)d_in[12];
  float* out = (float*)d_out;
  (void)n_in; (void)out_size; (void)ws_size;

  const int N = in_sizes[0] / CH;   // 50000
  const int E = in_sizes[1] / 2;    // 600000
  const int* src = eidx;
  const int* dst = eidx + E;

  // workspace carve-out (~105 MB + packed weights)
  char* ws = (char*)d_ws;
  size_t off = 0;
  auto carve = [&](size_t bytes) -> void* {
    void* p = ws + off;
    off += (bytes + 255) & ~(size_t)255;
    return p;
  };
  const size_t WPE = (size_t)KTILES * CTILES * 32 * 16;   // 49152 bf16 per layer
  float*  dinv = (float*)carve((size_t)N * 4);
  float*  ew   = (float*)carve((size_t)E * 4);
  float*  bufA = (float*)carve((size_t)N * CH * 4);
  float*  bufO = (float*)carve((size_t)N * CH * 4);
  float*  tx1  = (float*)carve((size_t)N * CH * 4);
  float*  tx2  = (float*)carve((size_t)N * CH * 4);
  float*  g    = (float*)carve((size_t)NUM_GRAPHS * CH * 4);
  __bf16* Wp[5];
  for (int l = 0; l < 5; ++l) Wp[l] = (__bf16*)carve(WPE * 2);

  const int NE = N * CH;
  const int TB = 256;

  // --- pack all 5 layer weights to bf16 fragment-major ---
  const float* Wsrc[5] = {W1, Wh + 0 * 3 * CH * CH, Wh + 1 * 3 * CH * CH,
                          Wh + 2 * 3 * CH * CH, W2};
  for (int l = 0; l < 5; ++l)
    pack_w_bf16<<<(int)((WPE + TB - 1) / TB), TB, 0, stream>>>(Wsrc[l], Wp[l]);

  // --- sym normalization: w = -d^-1/2 A d^-1/2 (self loops removed) ---
  zero_f32   <<<(N + TB - 1) / TB, TB, 0, stream>>>(dinv, N);
  deg_kernel <<<(E + TB - 1) / TB, TB, 0, stream>>>(src, dst, dinv, E);
  dinv_kernel<<<(N + TB - 1) / TB, TB, 0, stream>>>(dinv, N);
  ew_kernel  <<<(E + TB - 1) / TB, TB, 0, stream>>>(src, dst, dinv, ew, E);

  const int spmvBlocks = (E * 32 + TB - 1) / TB;
  const int ntiles     = (N + 15) / 16;          // 3125
  const int gemmBlocks = (ntiles + 7) / 8;       // 8 row-tiles (waves) per block

  auto layer = [&](const float* in, const __bf16* Wpk, const float* bias,
                   float* o, int act) {
    zero_f32<<<(NE + TB - 1) / TB, TB, 0, stream>>>(tx1, NE);
    spmv_scatter<<<spmvBlocks, TB, 0, stream>>>(in, src, dst, ew, tx1, E);
    zero_f32<<<(NE + TB - 1) / TB, TB, 0, stream>>>(tx2, NE);
    spmv_scatter<<<spmvBlocks, TB, 0, stream>>>(tx1, src, dst, ew, tx2, E);
    cheb_t2_kernel<<<(NE + TB - 1) / TB, TB, 0, stream>>>(tx2, in, NE);
    cheb_gemm_wmma<<<gemmBlocks, 256, 0, stream>>>(in, tx1, tx2, Wpk, bias, o,
                                                   ntiles, act);
  };

  layer(x,    Wp[0], b1,          bufO, 0);   // conv1 (no act)
  layer(bufO, Wp[1], bh + 0 * CH, bufA, 1);   // hidden 0 + SiLU
  layer(bufA, Wp[2], bh + 1 * CH, bufO, 1);   // hidden 1 + SiLU
  layer(bufO, Wp[3], bh + 2 * CH, bufA, 1);   // hidden 2 + SiLU
  layer(bufA, Wp[4], b2,          bufO, 0);   // conv out (no act)

  // --- pooling + output MLP ---
  zero_f32<<<(NUM_GRAPHS * CH + TB - 1) / TB, TB, 0, stream>>>(g, NUM_GRAPHS * CH);
  pool_kernel<<<(N * 32 + TB - 1) / TB, TB, 0, stream>>>(bufO, batc, g, N);
  mlp_kernel<<<1, 512, 0, stream>>>(g, P1, pb1, P2, pb2, out);
}
